// NeuroPredessor_66142496358602
// MI455X (gfx1250) — matrix operand order repaired and checked by reading, not compiled, not run on hardware
//
#include <hip/hip_runtime.h>
#include <cstdint>
#include <cstddef>

#define DIM 128
#define N_ROUNDS 120

typedef __attribute__((ext_vector_type(16))) __bf16 v16bf;
typedef __attribute__((ext_vector_type(8)))  float  v8f;

union ABfrag {
    v16bf    v;
    uint32_t u[8];
    uint4    q[2];
};

__device__ __forceinline__ int imin(int a, int b) { return a < b ? a : b; }
__device__ __forceinline__ float sigmoidf(float x) { return 1.0f / (1.0f + __expf(-x)); }

// ---- A fragment (16x32 bf16) from f32 row base rp = &A[m*ld + 8*laneHalf] ----
// Lane's fragment = two contiguous 8-float segments: [kb, kb+8) and [kb+16, kb+24)
// (each shifted by 8*laneHalf, folded into rp). 4x b128 loads + pack-convert.
__device__ __forceinline__ v16bf load_a_f32_frag(const float* __restrict__ rp, int kb)
{
    float4 x0 = *reinterpret_cast<const float4*>(rp + kb);
    float4 x1 = *reinterpret_cast<const float4*>(rp + kb + 4);
    float4 x2 = *reinterpret_cast<const float4*>(rp + kb + 16);
    float4 x3 = *reinterpret_cast<const float4*>(rp + kb + 20);
    ABfrag f;
    f.v[0]  = (__bf16)x0.x; f.v[1]  = (__bf16)x0.y; f.v[2]  = (__bf16)x0.z; f.v[3]  = (__bf16)x0.w;
    f.v[4]  = (__bf16)x1.x; f.v[5]  = (__bf16)x1.y; f.v[6]  = (__bf16)x1.z; f.v[7]  = (__bf16)x1.w;
    f.v[8]  = (__bf16)x2.x; f.v[9]  = (__bf16)x2.y; f.v[10] = (__bf16)x2.z; f.v[11] = (__bf16)x2.w;
    f.v[12] = (__bf16)x3.x; f.v[13] = (__bf16)x3.y; f.v[14] = (__bf16)x3.z; f.v[15] = (__bf16)x3.w;
    return f.v;
}

// ---- A fragment from LDS bf16 tile; tp = &tile[m*DIM + 8*laneHalf]; 2x ds b128 ----
__device__ __forceinline__ v16bf load_a_lds_frag(const __bf16* tp, int kb)
{
    ABfrag f;
    f.q[0] = *reinterpret_cast<const uint4*>(tp + kb);
    f.q[1] = *reinterpret_cast<const uint4*>(tp + kb + 16);
    return f.v;
}

// ---- B fragment from fragment-ready layout: frag blocks of 32 lanes x 32B ----
// Bl = B + lane*16 (bf16 units); fragment `frag` lives at +frag*512 elements.
__device__ __forceinline__ v16bf load_b_frag(const __bf16* __restrict__ Bl, int frag)
{
    const __bf16* p = Bl + (size_t)frag * 512;
    ABfrag f;
    f.q[0] = *reinterpret_cast<const uint4*>(p);
    f.q[1] = *reinterpret_cast<const uint4*>(p + 8);
    return f.v;
}

// =====================================================================
// Fused 3-layer MLP (relu, relu, linear), DIM=128, bf16 WMMA, f32 accum.
// Block = 128 threads = 4 waves; each wave owns a 16-row strip.
// =====================================================================
__global__ void __launch_bounds__(128)
mlp3_wmma_kernel(const float* __restrict__ x, int rows,
                 const __bf16* __restrict__ W1, const float* __restrict__ b1,
                 const __bf16* __restrict__ W2, const float* __restrict__ b2,
                 const __bf16* __restrict__ W3, const float* __restrict__ b3,
                 float* __restrict__ out)
{
    __shared__ __align__(16) __bf16 tiles[4][16 * DIM];
    const int wave = threadIdx.x >> 5;
    const int lane = threadIdx.x & 31;
    const int lh   = (lane >> 4) & 1;
    const int nn   = lane & 15;
    const int row0 = (blockIdx.x * 4 + wave) * 16;
    __bf16* tile = tiles[wave];

    int m_in = row0 + (lane & 15);
    if (m_in >= rows) m_in = rows - 1;
    const float*  arow = x + (size_t)m_in * DIM + lh * 8;          // A base (global f32)
    const __bf16* trow = tile + (lane & 15) * DIM + lh * 8;        // A base (LDS bf16)
    const __bf16* W1l = W1 + lane * 16;
    const __bf16* W2l = W2 + lane * 16;
    const __bf16* W3l = W3 + lane * 16;

    v8f acc[8];

    // ---------- layer 1: A from global f32 ----------
#pragma unroll
    for (int nt = 0; nt < 8; ++nt)
#pragma unroll
        for (int r = 0; r < 8; ++r) acc[nt][r] = 0.0f;
#pragma unroll
    for (int kk = 0; kk < 4; ++kk) {
        v16bf a = load_a_f32_frag(arow, kk * 32);
#pragma unroll
        for (int nt = 0; nt < 8; ++nt)
            acc[nt] = __builtin_amdgcn_wmma_f32_16x16x32_bf16(
                false, a, false, load_b_frag(W1l, nt * 4 + kk), (short)0, acc[nt], false, false);
    }
#pragma unroll
    for (int nt = 0; nt < 8; ++nt)
#pragma unroll
        for (int r = 0; r < 8; ++r) {
            int m = r + 8 * lh, n = nt * 16 + nn;
            float v = acc[nt][r] + b1[n];
            tile[m * DIM + n] = (__bf16)(v > 0.0f ? v : 0.0f);
        }
    __syncthreads();

    // ---------- layer 2: A from LDS bf16 ----------
#pragma unroll
    for (int nt = 0; nt < 8; ++nt)
#pragma unroll
        for (int r = 0; r < 8; ++r) acc[nt][r] = 0.0f;
#pragma unroll
    for (int kk = 0; kk < 4; ++kk) {
        v16bf a = load_a_lds_frag(trow, kk * 32);
#pragma unroll
        for (int nt = 0; nt < 8; ++nt)
            acc[nt] = __builtin_amdgcn_wmma_f32_16x16x32_bf16(
                false, a, false, load_b_frag(W2l, nt * 4 + kk), (short)0, acc[nt], false, false);
    }
    __syncthreads();
#pragma unroll
    for (int nt = 0; nt < 8; ++nt)
#pragma unroll
        for (int r = 0; r < 8; ++r) {
            int m = r + 8 * lh, n = nt * 16 + nn;
            float v = acc[nt][r] + b2[n];
            tile[m * DIM + n] = (__bf16)(v > 0.0f ? v : 0.0f);
        }
    __syncthreads();

    // ---------- layer 3: linear, store f32 ----------
#pragma unroll
    for (int nt = 0; nt < 8; ++nt)
#pragma unroll
        for (int r = 0; r < 8; ++r) acc[nt][r] = 0.0f;
#pragma unroll
    for (int kk = 0; kk < 4; ++kk) {
        v16bf a = load_a_lds_frag(trow, kk * 32);
#pragma unroll
        for (int nt = 0; nt < 8; ++nt)
            acc[nt] = __builtin_amdgcn_wmma_f32_16x16x32_bf16(
                false, a, false, load_b_frag(W3l, nt * 4 + kk), (short)0, acc[nt], false, false);
    }
#pragma unroll
    for (int nt = 0; nt < 8; ++nt)
#pragma unroll
        for (int r = 0; r < 8; ++r) {
            int m = row0 + r + 8 * lh, n = nt * 16 + nn;
            if (m < rows) out[(size_t)m * DIM + n] = acc[nt][r] + b3[n];
        }
}

// =====================================================================
// Fused LSTM step: gates = x@Wih^T + h@Whh^T + (bih+bhh); then i,f,g,o.
// Block = 256 threads = 8 waves, 16 rows x 512 gate cols; wave w owns
// gate columns [64w, 64w+64). In-place h/c update (own rows only).
// =====================================================================
__global__ void __launch_bounds__(256)
lstm_wmma_kernel(const float* __restrict__ x, float* __restrict__ h, float* __restrict__ c,
                 int rows,
                 const __bf16* __restrict__ Wih, const __bf16* __restrict__ Whh,
                 const float* __restrict__ bih, const float* __restrict__ bhh)
{
    __shared__ __align__(16) float gates[16 * 4 * DIM]; // 16 rows x 512 = 32KB
    const int wave  = threadIdx.x >> 5;
    const int lane  = threadIdx.x & 31;
    const int lh    = (lane >> 4) & 1;
    const int nn    = lane & 15;
    const int row0  = blockIdx.x * 16;
    const int ncol0 = wave * 64;

    int m_in = row0 + (lane & 15);
    if (m_in >= rows) m_in = rows - 1;
    const float* xrow = x + (size_t)m_in * DIM + lh * 8;
    const float* hrow = h + (size_t)m_in * DIM + lh * 8;
    // per-wave weight bases: fragment index = (wave*4 + j)*4 + kk -> fold wave in
    const __bf16* Wihl = Wih + lane * 16 + (size_t)wave * 16 * 512;
    const __bf16* Whhl = Whh + lane * 16 + (size_t)wave * 16 * 512;

    v8f acc[4];
#pragma unroll
    for (int j = 0; j < 4; ++j)
#pragma unroll
        for (int r = 0; r < 8; ++r) acc[j][r] = 0.0f;

#pragma unroll
    for (int kk = 0; kk < 4; ++kk) {
        v16bf a = load_a_f32_frag(xrow, kk * 32);
#pragma unroll
        for (int j = 0; j < 4; ++j)
            acc[j] = __builtin_amdgcn_wmma_f32_16x16x32_bf16(
                false, a, false, load_b_frag(Wihl, j * 4 + kk), (short)0, acc[j], false, false);
    }
#pragma unroll
    for (int kk = 0; kk < 4; ++kk) {
        v16bf a = load_a_f32_frag(hrow, kk * 32);
#pragma unroll
        for (int j = 0; j < 4; ++j)
            acc[j] = __builtin_amdgcn_wmma_f32_16x16x32_bf16(
                false, a, false, load_b_frag(Whhl, j * 4 + kk), (short)0, acc[j], false, false);
    }
#pragma unroll
    for (int j = 0; j < 4; ++j)
#pragma unroll
        for (int r = 0; r < 8; ++r) {
            int m = r + 8 * lh, n = ncol0 + j * 16 + nn;
            gates[m * (4 * DIM) + n] = acc[j][r] + bih[n] + bhh[n];
        }
    __syncthreads();

#pragma unroll
    for (int it = 0; it < 2; ++it) {
        int q = threadIdx.x + it * 256;          // 0..511 => (m, 4-wide d chunk)
        int m = q >> 5, d0 = (q & 31) << 2;
        int grow = row0 + m;
        if (grow < rows) {
            const float* g = gates + m * (4 * DIM);
            float4 vi = *reinterpret_cast<const float4*>(g + d0);
            float4 vf = *reinterpret_cast<const float4*>(g + DIM + d0);
            float4 vg = *reinterpret_cast<const float4*>(g + 2 * DIM + d0);
            float4 vo = *reinterpret_cast<const float4*>(g + 3 * DIM + d0);
            float4* cp = reinterpret_cast<float4*>(c + (size_t)grow * DIM + d0);
            float4* hp = reinterpret_cast<float4*>(h + (size_t)grow * DIM + d0);
            float4 cv = *cp;
            float4 cn, hn;
            cn.x = sigmoidf(vf.x) * cv.x + sigmoidf(vi.x) * tanhf(vg.x);
            cn.y = sigmoidf(vf.y) * cv.y + sigmoidf(vi.y) * tanhf(vg.y);
            cn.z = sigmoidf(vf.z) * cv.z + sigmoidf(vi.z) * tanhf(vg.z);
            cn.w = sigmoidf(vf.w) * cv.w + sigmoidf(vi.w) * tanhf(vg.w);
            hn.x = sigmoidf(vo.x) * tanhf(cn.x);
            hn.y = sigmoidf(vo.y) * tanhf(cn.y);
            hn.z = sigmoidf(vo.z) * tanhf(cn.z);
            hn.w = sigmoidf(vo.w) * tanhf(cn.w);
            *cp = cn;
            *hp = hn;
        }
    }
}

// =====================================================================
// SpMM as ordered gather-sum over CSR. Wave per row, float4 per lane.
// =====================================================================
__global__ void __launch_bounds__(256)
gather_rows_kernel(const float* __restrict__ src, const int* __restrict__ ptr,
                   const int* __restrict__ idx, int cap, int rows, float* __restrict__ dst)
{
    const int wave = threadIdx.x >> 5;
    const int lane = threadIdx.x & 31;
    const int row  = blockIdx.x * 8 + wave;
    if (row >= rows) return;
    int e0 = ptr[row];
    int e1 = imin(ptr[row + 1], cap);
    float4 s = {0.0f, 0.0f, 0.0f, 0.0f};
    for (int e = e0; e < e1; ++e) {
        if (e + 1 < e1)
            __builtin_prefetch(reinterpret_cast<const float4*>(src + (size_t)idx[e + 1] * DIM) + lane, 0, 1);
        float4 v = *(reinterpret_cast<const float4*>(src + (size_t)idx[e] * DIM) + lane);
        s.x += v.x; s.y += v.y; s.z += v.z; s.w += v.w;
    }
    *(reinterpret_cast<float4*>(dst + (size_t)row * DIM) + lane) = s;
}

// ---- CSR construction (deterministic, no atomics) ----
__global__ void __launch_bounds__(256)
count_kernel(const float* __restrict__ adj, int* __restrict__ ptr,
             int n_iter, size_t fixed_stride, size_t iter_stride)
{
    __shared__ int red[256];
    const int fi = blockIdx.x;
    const float* basep = adj + (size_t)fi * fixed_stride;
    const int chunk = (n_iter + 255) / 256;
    int t0 = threadIdx.x * chunk;
    int t1 = imin(t0 + chunk, n_iter);
    int cnt = 0;
    for (int t = t0; t < t1; ++t) cnt += (basep[(size_t)t * iter_stride] != 0.0f) ? 1 : 0;
    red[threadIdx.x] = cnt;
    __syncthreads();
    for (int s = 128; s > 0; s >>= 1) {
        if ((int)threadIdx.x < s) red[threadIdx.x] += red[threadIdx.x + s];
        __syncthreads();
    }
    if (threadIdx.x == 0) ptr[fi + 1] = red[0];
}

__global__ void prefix_kernel(int* ptr, int n)
{
    if (threadIdx.x == 0 && blockIdx.x == 0) {
        int run = 0;
        ptr[0] = 0;
        for (int i = 1; i <= n; ++i) { run += ptr[i]; ptr[i] = run; }
    }
}

__global__ void __launch_bounds__(256)
fill_kernel(const float* __restrict__ adj, const int* __restrict__ ptr,
            int* __restrict__ out_idx, int cap,
            int n_iter, size_t fixed_stride, size_t iter_stride)
{
    __shared__ int red[256];
    const int fi = blockIdx.x;
    const float* basep = adj + (size_t)fi * fixed_stride;
    const int chunk = (n_iter + 255) / 256;
    int t0 = threadIdx.x * chunk;
    int t1 = imin(t0 + chunk, n_iter);
    int cnt = 0;
    for (int t = t0; t < t1; ++t) cnt += (basep[(size_t)t * iter_stride] != 0.0f) ? 1 : 0;
    red[threadIdx.x] = cnt;
    __syncthreads();
    if (threadIdx.x == 0) {                 // ordered exclusive scan
        int run = 0;
        for (int i = 0; i < 256; ++i) { int v = red[i]; red[i] = run; run += v; }
    }
    __syncthreads();
    int ofs = ptr[fi] + red[threadIdx.x];
    for (int t = t0; t < t1; ++t)
        if (basep[(size_t)t * iter_stride] != 0.0f) {
            if (ofs < cap) out_idx[ofs] = t;
            ++ofs;
        }
}

// ---- weight fp32 -> bf16 fragment-ready layout ----
// B[k][n] = W[n][k]; fragment (nt, kk) is 32 lanes x 16 bf16 contiguous:
//   lane = ((krem>>3)&1)*16 + (n&15); dword e = ((krem&7)|((krem>>4)<<3))>>1; sub = k&1
__global__ void wtr_kernel(const float* __restrict__ W, __bf16* __restrict__ Wt, int O, int K)
{
    int idx = blockIdx.x * blockDim.x + threadIdx.x;
    if (idx >= O * K) return;
    int k = idx / O, n = idx - k * O;
    int nt = n >> 4, nn = n & 15;
    int kk = k >> 5, krem = k & 31;
    int lane = ((krem >> 3) & 1) * 16 + nn;
    int e = ((krem & 7) | ((krem >> 4) << 3)) >> 1;
    size_t dst = ((size_t)(nt * (K / 32) + kk) * 512) + lane * 16 + e * 2 + (k & 1);
    Wt[dst] = (__bf16)W[(size_t)n * K + k];
}

__global__ void init_kernel(const int* __restrict__ values,
                            const float* __restrict__ tW, const float* __restrict__ tb,
                            const float* __restrict__ fW, const float* __restrict__ fb,
                            float* __restrict__ h, float* __restrict__ c, int n_total)
{
    int idx = blockIdx.x * blockDim.x + threadIdx.x;
    if (idx >= n_total * DIM) return;
    int r = idx >> 7, d = idx & (DIM - 1);
    h[idx] = (values[r] == 1) ? (tW[d] + tb[d]) : (fW[d] + fb[d]);
    c[idx] = 0.0f;
}

// ---- vote head: relu(128x128) x2 then dot with (1,128); tiny, once ----
__global__ void __launch_bounds__(128)
vote_kernel(const float* __restrict__ hbuf, int n_nodes,
            const float* __restrict__ W1, const float* __restrict__ b1,
            const float* __restrict__ W2, const float* __restrict__ b2,
            const float* __restrict__ W3, const float* __restrict__ b3,
            float* __restrict__ out)
{
    __shared__ float s0[DIM], s1[DIM], red[DIM];
    const int row = blockIdx.x;
    const int t = threadIdx.x;
    const float* hr = hbuf + (size_t)(n_nodes + row) * DIM;
    s0[t] = hr[t];
    __syncthreads();
    float a = 0.0f;
    for (int k = 0; k < DIM; ++k) a += s0[k] * W1[t * DIM + k];
    a += b1[t]; a = a > 0.0f ? a : 0.0f;
    s1[t] = a;
    __syncthreads();
    a = 0.0f;
    for (int k = 0; k < DIM; ++k) a += s1[k] * W2[t * DIM + k];
    a += b2[t]; a = a > 0.0f ? a : 0.0f;
    red[t] = a * W3[t];
    __syncthreads();
    for (int s = 64; s > 0; s >>= 1) { if (t < s) red[t] += red[t + s]; __syncthreads(); }
    if (t == 0) out[row] = red[0] + b3[0];
}

extern "C" void kernel_launch(void* const* d_in, const int* in_sizes, int n_in,
                              void* d_out, int out_size, void* d_ws, size_t ws_size,
                              hipStream_t stream)
{
    (void)n_in; (void)out_size; (void)ws_size;

    const float* adj    = (const float*)d_in[0];
    const int*   values = (const int*)d_in[1];
    const int n_total = in_sizes[1];              // 9000
    const int n_nodes = in_sizes[0] / n_total;    // 8000
    const int n_vars  = n_total - n_nodes;        // 1000
    const int CAP = 2000000;                      // edge capacity per direction

    // ---- workspace carve ----
    char* base = (char*)d_ws;
    size_t off = 0;
    auto carve = [&](size_t bytes) -> void* {
        void* p = base + off;
        off = (off + bytes + 255) & ~(size_t)255;
        return p;
    };
    float* h   = (float*)carve((size_t)n_total * DIM * 4);
    float* cst = (float*)carve((size_t)n_total * DIM * 4);
    float* pre = (float*)carve((size_t)n_total * DIM * 4);
    float* agg = (float*)carve((size_t)n_total * DIM * 4);
    __bf16* cmW1 = (__bf16*)carve((size_t)DIM * DIM * 2);
    __bf16* cmW2 = (__bf16*)carve((size_t)DIM * DIM * 2);
    __bf16* cmW3 = (__bf16*)carve((size_t)DIM * DIM * 2);
    __bf16* pmW1 = (__bf16*)carve((size_t)DIM * DIM * 2);
    __bf16* pmW2 = (__bf16*)carve((size_t)DIM * DIM * 2);
    __bf16* pmW3 = (__bf16*)carve((size_t)DIM * DIM * 2);
    __bf16* vuWih = (__bf16*)carve((size_t)4 * DIM * DIM * 2);
    __bf16* vuWhh = (__bf16*)carve((size_t)4 * DIM * DIM * 2);
    __bf16* nuWih = (__bf16*)carve((size_t)4 * DIM * DIM * 2);
    __bf16* nuWhh = (__bf16*)carve((size_t)4 * DIM * DIM * 2);
    int* rowptr = (int*)carve((size_t)(n_nodes + 1) * 4);
    int* colptr = (int*)carve((size_t)(n_total + 1) * 4);
    int* colidx = (int*)carve((size_t)CAP * 4);
    int* rowidx = (int*)carve((size_t)CAP * 4);

    // ---- one-time setup (runs every launch; deterministic) ----
    auto wtr = [&](int i, __bf16* Wt, int O, int K) {
        int n = O * K;
        wtr_kernel<<<(n + 255) / 256, 256, 0, stream>>>((const float*)d_in[i], Wt, O, K);
    };
    wtr(7,  cmW1, DIM, DIM);  wtr(9,  cmW2, DIM, DIM);  wtr(11, cmW3, DIM, DIM);
    wtr(13, pmW1, DIM, DIM);  wtr(15, pmW2, DIM, DIM);  wtr(17, pmW3, DIM, DIM);
    wtr(25, vuWih, 4 * DIM, DIM);  wtr(26, vuWhh, 4 * DIM, DIM);
    wtr(29, nuWih, 4 * DIM, DIM);  wtr(30, nuWhh, 4 * DIM, DIM);

    count_kernel<<<n_nodes, 256, 0, stream>>>(adj, rowptr, n_total, (size_t)n_total, (size_t)1);
    prefix_kernel<<<1, 1, 0, stream>>>(rowptr, n_nodes);
    fill_kernel<<<n_nodes, 256, 0, stream>>>(adj, rowptr, colidx, CAP, n_total, (size_t)n_total, (size_t)1);
    count_kernel<<<n_total, 256, 0, stream>>>(adj, colptr, n_nodes, (size_t)1, (size_t)n_total);
    prefix_kernel<<<1, 1, 0, stream>>>(colptr, n_total);
    fill_kernel<<<n_total, 256, 0, stream>>>(adj, colptr, rowidx, CAP, n_nodes, (size_t)1, (size_t)n_total);

    int tot = n_total * DIM;
    init_kernel<<<(tot + 255) / 256, 256, 0, stream>>>(
        values, (const float*)d_in[3], (const float*)d_in[4],
        (const float*)d_in[5], (const float*)d_in[6], h, cst, n_total);

    const float* cm_b1 = (const float*)d_in[8],  *cm_b2 = (const float*)d_in[10], *cm_b3 = (const float*)d_in[12];
    const float* pm_b1 = (const float*)d_in[14], *pm_b2 = (const float*)d_in[16], *pm_b3 = (const float*)d_in[18];
    const float* vu_bih = (const float*)d_in[27], *vu_bhh = (const float*)d_in[28];
    const float* nu_bih = (const float*)d_in[31], *nu_bhh = (const float*)d_in[32];

    // ---- 120 message-passing rounds ----
    for (int t = 0; t < N_ROUNDS; ++t) {
        mlp3_wmma_kernel<<<(n_total + 63) / 64, 128, 0, stream>>>(
            h, n_total, cmW1, cm_b1, cmW2, cm_b2, cmW3, cm_b3, pre);
        gather_rows_kernel<<<(n_nodes + 7) / 8, 256, 0, stream>>>(pre, rowptr, colidx, CAP, n_nodes, agg);
        lstm_wmma_kernel<<<(n_nodes + 15) / 16, 256, 0, stream>>>(
            agg, h, cst, n_nodes, vuWih, vuWhh, vu_bih, vu_bhh);
        mlp3_wmma_kernel<<<(n_nodes + 63) / 64, 128, 0, stream>>>(
            h, n_nodes, pmW1, pm_b1, pmW2, pm_b2, pmW3, pm_b3, pre);
        gather_rows_kernel<<<(n_total + 7) / 8, 256, 0, stream>>>(pre, colptr, rowidx, CAP, n_total, agg);
        lstm_wmma_kernel<<<(n_total + 15) / 16, 256, 0, stream>>>(
            agg, h, cst, n_total, nuWih, nuWhh, nu_bih, nu_bhh);
    }

    vote_kernel<<<n_vars, DIM, 0, stream>>>(
        h, n_nodes,
        (const float*)d_in[19], (const float*)d_in[20],
        (const float*)d_in[21], (const float*)d_in[22],
        (const float*)d_in[23], (const float*)d_in[24],
        (float*)d_out);
}
